// InfoGraphEncoder_6382321402424
// MI455X (gfx1250) — matrix operand order repaired and checked by reading, not compiled, not run on hardware
//
#include <hip/hip_runtime.h>
#include <hip/hip_bf16.h>

typedef _Float16 h16;
typedef _Float16 v16h __attribute__((ext_vector_type(16)));
typedef _Float16 v8h  __attribute__((ext_vector_type(8)));
typedef float    v8f  __attribute__((ext_vector_type(8)));

#define N_NODES 65536
#define N_EDGES 131072
#define N_GRAPH 2048

__device__ inline v8f wmma_f16(v16h a, v16h b, v8f c) {
  return __builtin_amdgcn_wmma_f32_16x16x32_f16(false, a, false, b, (short)0, c, false, false);
}
__device__ inline v16h combine(v8h lo, v8h hi) {
  union { v16h v; v8h h[2]; } u; u.h[0] = lo; u.h[1] = hi; return u.v;
}
__device__ inline float sigmoidf(float x) { return 1.f / (1.f + __expf(-x)); }
__device__ inline unsigned f2ord(float x) {
  unsigned u = __float_as_uint(x);
  return (u & 0x80000000u) ? ~u : (u | 0x80000000u);
}
__device__ inline float ord2f(unsigned u) {
  return (u & 0x80000000u) ? __uint_as_float(u ^ 0x80000000u) : __uint_as_float(~u);
}

// ---------------- weight prep: f32 -> f16, transposed/reordered for WMMA B frags ----
__global__ void k_prep(const float* lin0_w, const float* mlp_w1, const float* mlp_w2,
                       const float* mlp_b2, const float* gwih, const float* gwhh,
                       h16* l0wt, h16* w1t, h16* Wt, h16* B2t, h16* giht, h16* ghht) {
  int i = blockIdx.x * blockDim.x + threadIdx.x;
  if (i < 64 * 96) { int f = i / 96, k = i % 96;
    l0wt[f * 96 + k] = (h16)(k < 75 ? lin0_w[k * 64 + f] : 0.f); return; }
  i -= 64 * 96;
  if (i < 128 * 32) { int f = i / 32, k = i % 32;
    w1t[f * 32 + k] = (h16)(k < 16 ? mlp_w1[k * 128 + f] : 0.f); return; }
  i -= 128 * 32;
  if (i < 64 * 8192) { int f = i / 8192, kd = i % 8192; int k = kd >> 6, d = kd & 63;
    Wt[(size_t)f * 8192 + kd] = (h16)mlp_w2[(size_t)k * 4096 + d * 64 + f]; return; }
  i -= 64 * 8192;
  if (i < 64 * 64) { int f = i / 64, d = i % 64;
    B2t[f * 64 + d] = (h16)mlp_b2[d * 64 + f]; return; }
  i -= 64 * 64;
  if (i < 192 * 64) { giht[i] = (h16)gwih[i]; return; }
  i -= 192 * 64;
  if (i < 192 * 64) { ghht[i] = (h16)gwhh[i]; }
}

__global__ void k_zero(float* p, long n) {
  long i = (long)blockIdx.x * blockDim.x + threadIdx.x;
  if (i < n) p[i] = 0.f;
}

__global__ void k_counts(const int* dst, float* counts) {
  int e = blockIdx.x * blockDim.x + threadIdx.x;
  if (e < N_EDGES) atomicAdd(&counts[dst[e]], 1.0f);
}

// ---------------- lin0: relu(node_features @ W0 + b0), N x 96(pad) x 64 ------------
__global__ void k_lin0(const float* nf, const h16* l0wt, const float* b0, float* h) {
  int lane = threadIdx.x & 31, wv = threadIdx.x >> 5;
  int row0 = (blockIdx.x * 8 + wv) * 16;
  int m = row0 + (lane & 15);
  int off = 16 * (lane >> 4);
  const float* rp = nf + (size_t)m * 75;
  v8f acc[4] = {};
#pragma unroll
  for (int kc = 0; kc < 3; ++kc) {
    int k0 = kc * 32;
    v16h a;
#pragma unroll
    for (int j = 0; j < 8; ++j) {
      // branchless boundary handling: clamp address, mask value (-> v_cndmask, no divergence)
      int k = k0 + off + j;
      float v = rp[k < 75 ? k : 0];
      a[j] = (h16)(k < 75 ? v : 0.f);
      int k2 = k0 + off + 8 + j;
      float v2 = rp[k2 < 75 ? k2 : 0];
      a[8 + j] = (h16)(k2 < 75 ? v2 : 0.f);
    }
#pragma unroll
    for (int t = 0; t < 4; ++t) {
      v16h b = *(const v16h*)(l0wt + (size_t)(t * 16 + (lane & 15)) * 96 + k0 + 16 * (lane >> 4));
      acc[t] = wmma_f16(a, b, acc[t]);
    }
  }
#pragma unroll
  for (int t = 0; t < 4; ++t)
#pragma unroll
    for (int r = 0; r < 8; ++r) {
      int rr = row0 + r + 8 * (lane >> 4);
      int f = t * 16 + (lane & 15);
      float v = acc[t][r] + b0[f];
      h[(size_t)rr * 64 + f] = v > 0.f ? v : 0.f;
    }
}

// ---------------- edge MLP layer1: h1 = relu(ef @ W1 + b1), E x 32(pad) x 128 ------
__global__ void k_mlp1(const float* ef, const h16* w1t, const float* b1, h16* h1) {
  int lane = threadIdx.x & 31, wv = threadIdx.x >> 5;
  int row0 = (blockIdx.x * 8 + wv) * 16;
  int m = row0 + (lane & 15);
  int off = 8 * (lane >> 4);
  const float* rp = ef + (size_t)m * 16;
  v16h a;
#pragma unroll
  for (int j = 0; j < 8; ++j) { a[j] = (h16)(lane < 16 ? rp[j] : 0.f); a[8 + j] = (h16)(lane < 16 ? rp[8 + j] : 0.f); }
#pragma unroll
  for (int t = 0; t < 8; ++t) {
    v8f acc = {};
    v16h b = *(const v16h*)(w1t + (size_t)(t * 16 + (lane & 15)) * 32 + 16 * (lane >> 4));
    acc = wmma_f16(a, b, acc);
#pragma unroll
    for (int r = 0; r < 8; ++r) {
      int rr = row0 + r + 8 * (lane >> 4);
      int f = t * 16 + (lane & 15);
      float v = acc[r] + b1[f];
      h1[(size_t)rr * 128 + f] = (h16)(v > 0.f ? v : 0.f);
    }
  }
}

// ---------------- NNConv message pass: msg = Z @ W2'  (A synthesized in regs) ------
// Z[e, k*64+d] = h1[e,k] * x[e,d];  msg = Z @ W2' + x @ B2  (E x 8192 x 64 GEMM)
#define XS 72
#define HS 144
__global__ void k_msg(const float* h, const h16* h1, const int* src, const int* dst,
                      const h16* Wt, const h16* B2t, float* agg) {
  __shared__ __attribute__((aligned(32))) h16 xs[16 * XS];
  __shared__ __attribute__((aligned(32))) h16 hl[16 * HS];
  __shared__ int di[16];
  int t = threadIdx.x;
  int e0 = blockIdx.x * 16;
  { // gather x = h[src] -> f16 LDS
    int r = t >> 3, c = (t & 7) * 8;
    const float* rp = h + (size_t)src[e0 + r] * 64 + c;
#pragma unroll
    for (int j = 0; j < 8; ++j) xs[r * XS + c + j] = (h16)rp[j];
  }
  { // h1 tile 16x128 f16
    int r = t >> 3, c = (t & 7) * 16;
    *(v16h*)&hl[r * HS + c] = *(const v16h*)(h1 + (size_t)(e0 + r) * 128 + c);
  }
  if (t < 16) di[t] = dst[e0 + t];
  __syncthreads();

  int lane = t & 31, wv = t >> 5;   // wave = one 16-col output tile
  int er = lane & 15;
  int off = 16 * (lane >> 4);
  int fb = wv * 16 + (lane & 15);
  // x fragments are invariant over all 128 k-steps: hoist into registers
  v8h x0lo = *(const v8h*)&xs[er * XS + off];
  v8h x0hi = *(const v8h*)&xs[er * XS + off + 8];
  v8h x1lo = *(const v8h*)&xs[er * XS + 32 + off];
  v8h x1hi = *(const v8h*)&xs[er * XS + 32 + off + 8];
  const h16* wrow = Wt + (size_t)fb * 8192 + 16 * (lane >> 4);

  v8f acc = {};
  // bias term: x @ B2  (K = 64, two chunks)
  {
    const h16* brow = B2t + (size_t)fb * 64 + 16 * (lane >> 4);
    acc = wmma_f16(combine(x0lo, x0hi), *(const v16h*)(brow), acc);
    acc = wmma_f16(combine(x1lo, x1hi), *(const v16h*)(brow + 32), acc);
  }
  // main GEMM: for each k, two d0 chunks -> 2 WMMAs; unroll x4 for latency hiding
#pragma unroll 4
  for (int k = 0; k < 128; ++k) {
    h16 hv = hl[er * HS + k];
    v16h b0 = *(const v16h*)(wrow + (size_t)k * 64);
    v16h b1 = *(const v16h*)(wrow + (size_t)k * 64 + 32);
    acc = wmma_f16(combine(x0lo * hv, x0hi * hv), b0, acc);
    acc = wmma_f16(combine(x1lo * hv, x1hi * hv), b1, acc);
  }
  // scatter-add to destination nodes
#pragma unroll
  for (int r = 0; r < 8; ++r) {
    int node = di[r + 8 * (lane >> 4)];
    atomicAdd(&agg[(size_t)node * 64 + fb], acc[r]);
  }
}

// ---------------- GRU step: gates = m@Wih^T + h@Whh^T; n-gate kept split -----------
__global__ void k_gru(const float* agg, const float* counts, const float* cbias,
                      const h16* giht, const h16* ghht,
                      const float* bih, const float* bhh, float* h) {
  __shared__ __attribute__((aligned(32))) h16 mt[16 * XS];
  __shared__ __attribute__((aligned(32))) h16 ht[16 * XS];
  __shared__ float gs[16 * 128];
  __shared__ float gin[16 * 64];
  __shared__ float ghn[16 * 64];
  int t = threadIdx.x;
  int row0 = blockIdx.x * 16;
  { // m = relu(agg/count + conv_bias); stage m,h in f16 LDS
    int r = t >> 3, c = (t & 7) * 8;
    int node = row0 + r;
    float cnt = counts[node]; cnt = cnt < 1.f ? 1.f : cnt;
#pragma unroll
    for (int j = 0; j < 8; ++j) {
      float v = agg[(size_t)node * 64 + c + j] / cnt + cbias[c + j];
      mt[r * XS + c + j] = (h16)(v > 0.f ? v : 0.f);
      ht[r * XS + c + j] = (h16)h[(size_t)node * 64 + c + j];
    }
  }
  __syncthreads();
  int lane = t & 31, wv = t >> 5;
  int er = lane & 15, off = 16 * (lane >> 4);
  v16h am[2], ah[2];
#pragma unroll
  for (int kc = 0; kc < 2; ++kc) {
    am[kc] = combine(*(const v8h*)&mt[er * XS + kc * 32 + off],
                     *(const v8h*)&mt[er * XS + kc * 32 + off + 8]);
    ah[kc] = combine(*(const v8h*)&ht[er * XS + kc * 32 + off],
                     *(const v8h*)&ht[er * XS + kc * 32 + off + 8]);
  }
#pragma unroll
  for (int ti = 0; ti < 3; ++ti) {
    int tc = wv + ti * 4;
    int f = tc * 16 + (lane & 15);
    if (tc < 8) {          // r,z gates: gi+gh summed in one accumulator
      v8f acc = {};
#pragma unroll
      for (int kc = 0; kc < 2; ++kc) {
        acc = wmma_f16(am[kc], *(const v16h*)(giht + (size_t)f * 64 + kc * 32 + 16 * (lane >> 4)), acc);
        acc = wmma_f16(ah[kc], *(const v16h*)(ghht + (size_t)f * 64 + kc * 32 + 16 * (lane >> 4)), acc);
      }
#pragma unroll
      for (int r = 0; r < 8; ++r) gs[(r + 8 * (lane >> 4)) * 128 + f] = acc[r];
    } else {               // n gate: keep gi_n / gh_n separate
      v8f ai = {}, ag = {};
#pragma unroll
      for (int kc = 0; kc < 2; ++kc) {
        ai = wmma_f16(am[kc], *(const v16h*)(giht + (size_t)f * 64 + kc * 32 + 16 * (lane >> 4)), ai);
        ag = wmma_f16(ah[kc], *(const v16h*)(ghht + (size_t)f * 64 + kc * 32 + 16 * (lane >> 4)), ag);
      }
      int fn = f - 128;
#pragma unroll
      for (int r = 0; r < 8; ++r) {
        int mr = r + 8 * (lane >> 4);
        gin[mr * 64 + fn] = ai[r]; ghn[mr * 64 + fn] = ag[r];
      }
    }
  }
  __syncthreads();
  { // elementwise GRU update
    int r = t >> 3, c0 = (t & 7) * 8;
    int node = row0 + r;
#pragma unroll
    for (int j = 0; j < 8; ++j) {
      int c = c0 + j;
      float rg = sigmoidf(gs[r * 128 + c] + bih[c] + bhh[c]);
      float zg = sigmoidf(gs[r * 128 + 64 + c] + bih[64 + c] + bhh[64 + c]);
      float ng = tanhf(gin[r * 64 + c] + bih[128 + c] + rg * (ghn[r * 64 + c] + bhh[128 + c]));
      float hv = h[(size_t)node * 64 + c];
      h[(size_t)node * 64 + c] = (1.f - zg) * ng + zg * hv;
    }
  }
}

// ---------------- Set2Set (B=2048, tiny — VALU) ------------------------------------
__global__ void k_lstm(const float* qstar, const float* hs_in, float* cs, float* hs_out,
                       const float* wih, const float* whh, const float* bih, const float* bhh) {
  int i = blockIdx.x * blockDim.x + threadIdx.x;   // B*64
  int b = i >> 6, j = i & 63;
  const float* q = qstar + (size_t)b * 128;
  const float* hr = hs_in + (size_t)b * 64;
  float g[4];
#pragma unroll
  for (int gi = 0; gi < 4; ++gi) {
    int row = gi * 64 + j;
    const float* wi = wih + (size_t)row * 128;
    const float* wh = whh + (size_t)row * 64;
    float s = bih[row] + bhh[row];
    for (int k = 0; k < 128; ++k) s += q[k] * wi[k];
    for (int k = 0; k < 64; ++k)  s += hr[k] * wh[k];
    g[gi] = s;
  }
  float ig = sigmoidf(g[0]), fg = sigmoidf(g[1]), gg = tanhf(g[2]), og = sigmoidf(g[3]);
  float c = fg * cs[i] + ig * gg;
  cs[i] = c;
  hs_out[i] = og * tanhf(c);
}

__global__ void k_escore(const float* h, const float* q, const int* gidx,
                         float* esc, unsigned* emax) {
  int i = blockIdx.x * blockDim.x + threadIdx.x;   // N
  int g = gidx[i];
  const float* hr = h + (size_t)i * 64;
  const float* qr = q + (size_t)g * 64;
  float s = 0.f;
  for (int k = 0; k < 64; ++k) s += hr[k] * qr[k];
  esc[i] = s;
  atomicMax(&emax[g], f2ord(s));
}

__global__ void k_attn(const float* h, const float* esc, const unsigned* emax,
                       const int* gidx, float* asum, float* rvec) {
  int i = blockIdx.x * blockDim.x + threadIdx.x;   // N
  int g = gidx[i];
  float a = __expf(esc[i] - ord2f(emax[g]));
  atomicAdd(&asum[g], a);
  const float* hr = h + (size_t)i * 64;
  for (int k = 0; k < 64; ++k) atomicAdd(&rvec[(size_t)g * 64 + k], a * hr[k]);
}

__global__ void k_qstar(const float* q, const float* rvec, const float* asum, float* qstar) {
  int i = blockIdx.x * blockDim.x + threadIdx.x;   // B*64
  int b = i >> 6, j = i & 63;
  qstar[(size_t)b * 128 + j] = q[i];
  qstar[(size_t)b * 128 + 64 + j] = rvec[i] / (asum[b] + 1e-16f);
}

__global__ void k_output(const float* qstar, const float* h, float* out) {
  long i = (long)blockIdx.x * blockDim.x + threadIdx.x;
  const long nq = (long)N_GRAPH * 128;
  const long nf = (long)N_NODES * 64;
  if (i < nq) out[i] = qstar[i];
  else if (i < nq + nf) out[i] = h[i - nq];
}

extern "C" void kernel_launch(void* const* d_in, const int* in_sizes, int n_in,
                              void* d_out, int out_size, void* d_ws, size_t ws_size,
                              hipStream_t stream) {
  (void)in_sizes; (void)n_in; (void)out_size; (void)ws_size;
  const float* nf    = (const float*)d_in[0];
  const float* ef    = (const float*)d_in[1];
  const int*   eidx  = (const int*)d_in[2];
  const int*   gidx  = (const int*)d_in[3];
  const float* lin0w = (const float*)d_in[4];
  const float* lin0b = (const float*)d_in[5];
  const float* mw1   = (const float*)d_in[6];
  const float* mb1   = (const float*)d_in[7];
  const float* mw2   = (const float*)d_in[8];
  const float* mb2   = (const float*)d_in[9];
  const float* cbias = (const float*)d_in[10];
  const float* gwih  = (const float*)d_in[11];
  const float* gwhh  = (const float*)d_in[12];
  const float* gbih  = (const float*)d_in[13];
  const float* gbhh  = (const float*)d_in[14];
  const float* lwih  = (const float*)d_in[15];
  const float* lwhh  = (const float*)d_in[16];
  const float* lbih  = (const float*)d_in[17];
  const float* lbhh  = (const float*)d_in[18];
  const int* src = eidx;
  const int* dst = eidx + N_EDGES;

  char* wsb = (char*)d_ws;
  size_t cur = 0;
  auto alloc = [&](size_t bytes) -> void* {
    void* p = wsb + cur; cur += (bytes + 255) & ~(size_t)255; return p;
  };
  float* h      = (float*)alloc((size_t)N_NODES * 64 * 4);
  float* agg    = (float*)alloc((size_t)N_NODES * 64 * 4);
  h16*   h1     = (h16*)  alloc((size_t)N_EDGES * 128 * 2);
  h16*   Wt     = (h16*)  alloc((size_t)64 * 8192 * 2);
  h16*   B2t    = (h16*)  alloc(64 * 64 * 2);
  h16*   l0wt   = (h16*)  alloc(64 * 96 * 2);
  h16*   w1t    = (h16*)  alloc(128 * 32 * 2);
  h16*   giht   = (h16*)  alloc(192 * 64 * 2);
  h16*   ghht   = (h16*)  alloc(192 * 64 * 2);
  float* counts = (float*)alloc((size_t)N_NODES * 4);
  float* esc    = (float*)alloc((size_t)N_NODES * 4);
  unsigned* emax= (unsigned*)alloc(N_GRAPH * 4);
  float* asum   = (float*)alloc(N_GRAPH * 4);
  float* qstar  = (float*)alloc((size_t)N_GRAPH * 128 * 4);
  float* hsA    = (float*)alloc((size_t)N_GRAPH * 64 * 4);
  float* hsB    = (float*)alloc((size_t)N_GRAPH * 64 * 4);
  float* cs     = (float*)alloc((size_t)N_GRAPH * 64 * 4);
  float* rvec   = (float*)alloc((size_t)N_GRAPH * 64 * 4);

  // weight prep
  {
    int total = 64*96 + 128*32 + 64*8192 + 64*64 + 192*64 + 192*64;
    k_prep<<<(total + 255) / 256, 256, 0, stream>>>(lin0w, mw1, mw2, mb2, gwih, gwhh,
                                                    l0wt, w1t, Wt, B2t, giht, ghht);
  }
  // in-degree counts
  k_zero<<<(N_NODES + 255) / 256, 256, 0, stream>>>(counts, N_NODES);
  k_counts<<<N_EDGES / 256, 256, 0, stream>>>(dst, counts);
  // lin0 + edge MLP layer 1
  k_lin0<<<N_NODES / 128, 256, 0, stream>>>(nf, l0wt, lin0b, h);
  k_mlp1<<<N_EDGES / 128, 256, 0, stream>>>(ef, w1t, mb1, h1);
  // 3 rounds of NNConv + GRU
  for (int it = 0; it < 3; ++it) {
    k_zero<<<((long)N_NODES * 64 + 255) / 256, 256, 0, stream>>>(agg, (long)N_NODES * 64);
    k_msg<<<N_EDGES / 16, 128, 0, stream>>>(h, h1, src, dst, Wt, B2t, agg);
    k_gru<<<N_NODES / 16, 128, 0, stream>>>(agg, counts, cbias, giht, ghht, gbih, gbhh, h);
  }
  // Set2Set
  k_zero<<<((long)N_GRAPH * 128 + 255) / 256, 256, 0, stream>>>(qstar, (long)N_GRAPH * 128);
  k_zero<<<((long)N_GRAPH * 64 + 255) / 256, 256, 0, stream>>>(hsA, (long)N_GRAPH * 64);
  k_zero<<<((long)N_GRAPH * 64 + 255) / 256, 256, 0, stream>>>(cs, (long)N_GRAPH * 64);
  for (int it = 0; it < 3; ++it) {
    float* hin  = (it & 1) ? hsB : hsA;
    float* hout = (it & 1) ? hsA : hsB;
    k_lstm<<<(N_GRAPH * 64) / 256, 256, 0, stream>>>(qstar, hin, cs, hout, lwih, lwhh, lbih, lbhh);
    k_zero<<<(N_GRAPH + 255) / 256, 256, 0, stream>>>((float*)emax, N_GRAPH);
    k_zero<<<(N_GRAPH + 255) / 256, 256, 0, stream>>>(asum, N_GRAPH);
    k_zero<<<((long)N_GRAPH * 64 + 255) / 256, 256, 0, stream>>>(rvec, (long)N_GRAPH * 64);
    k_escore<<<N_NODES / 256, 256, 0, stream>>>(h, hout, gidx, esc, emax);
    k_attn<<<N_NODES / 256, 256, 0, stream>>>(h, esc, emax, gidx, asum, rvec);
    k_qstar<<<(N_GRAPH * 64) / 256, 256, 0, stream>>>(hout, rvec, asum, qstar);
  }
  // outputs: q_star then feat_map
  {
    long total = (long)N_GRAPH * 128 + (long)N_NODES * 64;
    k_output<<<(total + 255) / 256, 256, 0, stream>>>(qstar, h, (float*)d_out);
  }
}
